// EdgePropertyPredictionModel2_48928267436272
// MI455X (gfx1250) — matrix operand order repaired and checked by reading, not compiled, not run on hardware
//
#include <hip/hip_runtime.h>
#include <hip/hip_bf16.h>
#include <stdint.h>

#define N_NODES 100000
#define N_EDGES 600000
#define HID     128
#define HEADS   16
#define DHD     8
#define LAYERS  4
#define FF_DIM  256
#define BN_EPS  1e-5f
#define SLOPE   0.2f

typedef __bf16 bf16;
typedef __attribute__((ext_vector_type(16))) __bf16 bf16x16;
typedef __attribute__((ext_vector_type(8)))  __bf16 bf16x8;
typedef __attribute__((ext_vector_type(8)))  float  f32x8;

// ---------------- helpers ----------------

__device__ __forceinline__ bf16 to_bf16(float f) {
  unsigned u = __float_as_uint(f);
  unsigned r = (u + 0x7FFFu + ((u >> 16) & 1u)) >> 16;  // round-to-nearest-even
  union { unsigned short s; bf16 b; } c;
  c.s = (unsigned short)r;
  return c.b;
}

// order-preserving float <-> uint encoding (for atomic segment-max)
__device__ __forceinline__ unsigned fenc(float f) {
  unsigned u = __float_as_uint(f);
  return (u & 0x80000000u) ? ~u : (u | 0x80000000u);
}
__device__ __forceinline__ float fdec(unsigned e) {
  unsigned u = (e & 0x80000000u) ? (e ^ 0x80000000u) : ~e;
  return __uint_as_float(u);
}

// ---------------- WMMA fragment loaders (CDNA5 16x16x32 bf16 layouts) ----------------

// A: [rows x K] row-major bf16. Lane L holds row (rowBase + (L&15)).
// elements 0..7  -> K = kg+0..kg+7, elements 8..15 -> K = kg+16..kg+23, kg = (L>>4)*8.
__device__ __forceinline__ bf16x16 load_a_frag(const bf16* __restrict__ A, int lda,
                                               int rowBase, int kBase, int lane) {
  int m  = rowBase + (lane & 15);
  int kg = (lane >> 4) << 3;               // 0 or 8
  const bf16* p = A + (size_t)m * lda + kBase + kg;
  union { bf16x16 v; bf16x8 h[2]; } u;
  u.h[0] = *(const bf16x8*)(p);            // K = kg .. kg+7   (16B load)
  u.h[1] = *(const bf16x8*)(p + 16);       // K = kg+16..kg+23 (16B load)
  return u.v;
}

// B pre-packed into fragment order: element e of lane (n = col + L&15, kg = L>>4)
// lives at Bp[ ((kt*Ncols + n)*2 + kg)*16 + e ],  kt = kBase/32.
// -> two contiguous 16B loads per lane.
__device__ __forceinline__ bf16x16 load_b_frag(const bf16* __restrict__ Bp, int Ncols,
                                               int colBase, int kBase, int lane) {
  int n  = colBase + (lane & 15);
  int kg = lane >> 4;
  const bf16* p = Bp + ((size_t)((kBase >> 5) * Ncols + n) * 2 + kg) * 16;
  union { bf16x16 v; bf16x8 h[2]; } u;
  u.h[0] = *(const bf16x8*)(p);
  u.h[1] = *(const bf16x8*)(p + 8);
  return u.v;
}

__device__ __forceinline__ f32x8 wmma_bf16(bf16x16 a, bf16x16 b, f32x8 c) {
  return __builtin_amdgcn_wmma_f32_16x16x32_bf16(
      /*neg_a=*/false, a, /*neg_b=*/false, b,
      /*c_mod=*/(short)0, c, /*reuse_a=*/false, /*reuse_b=*/false);
}

// ---------------- generic WMMA GEMM ----------------
// out[M x Ncols] = A[M x K] @ B[K x Ncols] (+ bias[col]) (+ res) (relu?)
// grid: (M/32, Ncols/128); block: 256 threads = 8 waves.
// Each wave computes a 16x32 tile (A fragment reused for two WMMAs).
template <int K>
__global__ void __launch_bounds__(256)
gemm_wmma(const bf16* __restrict__ A, const bf16* __restrict__ Bp,
          const float* __restrict__ bias, const float* __restrict__ res,
          float* __restrict__ outF, bf16* __restrict__ outB,
          int Ncols, int relu)
{
  int lane = threadIdx.x & 31;
  int wave = threadIdx.x >> 5;
  int row  = blockIdx.x * 32 + (wave >> 2) * 16;
  int col  = blockIdx.y * 128 + (wave & 3) * 32;

  f32x8 acc0 = {0.f, 0.f, 0.f, 0.f, 0.f, 0.f, 0.f, 0.f};
  f32x8 acc1 = acc0;
#pragma unroll
  for (int k = 0; k < K; k += 32) {
    bf16x16 a  = load_a_frag(A, K, row, k, lane);
    bf16x16 b0 = load_b_frag(Bp, Ncols, col, k, lane);
    bf16x16 b1 = load_b_frag(Bp, Ncols, col + 16, k, lane);
    acc0 = wmma_bf16(a, b0, acc0);
    acc1 = wmma_bf16(a, b1, acc1);
  }

  // C/D layout: element v -> M = row + v + (lane>>4)*8, N = col + (lane&15)
  int n  = col + (lane & 15);
  int mh = row + ((lane >> 4) << 3);
  float bv0 = bias ? bias[n] : 0.f;
  float bv1 = bias ? bias[n + 16] : 0.f;
#pragma unroll
  for (int v = 0; v < 8; ++v) {
    size_t idx = (size_t)(mh + v) * Ncols + n;
    float val0 = acc0[v] + bv0;
    float val1 = acc1[v] + bv1;
    if (res)  { val0 += res[idx]; val1 += res[idx + 16]; }
    if (relu) { val0 = fmaxf(val0, 0.f); val1 = fmaxf(val1, 0.f); }
    if (outF) { outF[idx] = val0; outF[idx + 16] = val1; }
    if (outB) { outB[idx] = to_bf16(val0); outB[idx + 16] = to_bf16(val1); }
  }
}

// Decoder GEMM: hc[N,640] @ W1[640,128] with hc stored as 5 chunks of [N,128] bf16.
// grid: N/32; block 256 = 8 waves; wave computes 16x32.
__global__ void __launch_bounds__(256)
gemm_dec_wmma(const bf16* __restrict__ xsB, const bf16* __restrict__ W1p,
              const float* __restrict__ b1, float* __restrict__ outF)
{
  int lane = threadIdx.x & 31;
  int wave = threadIdx.x >> 5;
  int row  = blockIdx.x * 32 + (wave >> 2) * 16;
  int col  = (wave & 3) * 32;

  f32x8 acc0 = {0.f, 0.f, 0.f, 0.f, 0.f, 0.f, 0.f, 0.f};
  f32x8 acc1 = acc0;
#pragma unroll
  for (int chunk = 0; chunk < LAYERS + 1; ++chunk) {
    const bf16* A = xsB + (size_t)chunk * N_NODES * HID;
    const bf16* B = W1p + (size_t)chunk * HID * HID;
#pragma unroll
    for (int k = 0; k < HID; k += 32) {
      bf16x16 a = load_a_frag(A, HID, row, k, lane);
      acc0 = wmma_bf16(a, load_b_frag(B, HID, col, k, lane), acc0);
      acc1 = wmma_bf16(a, load_b_frag(B, HID, col + 16, k, lane), acc1);
    }
  }
  int n  = col + (lane & 15);
  int mh = row + ((lane >> 4) << 3);
  float bv0 = b1[n], bv1 = b1[n + 16];
#pragma unroll
  for (int v = 0; v < 8; ++v) {
    size_t idx = (size_t)(mh + v) * HID + n;
    outF[idx]      = fmaxf(acc0[v] + bv0, 0.f);
    outF[idx + 16] = fmaxf(acc1[v] + bv1, 0.f);
  }
}

// ---------------- small kernels ----------------

// Pack a [K x Ncols] f32 weight matrix into bf16 WMMA-fragment order.
// grid.y = batch index (contiguous K*Ncols matrices).
__global__ void pack_b_kernel(const float* __restrict__ in, bf16* __restrict__ out,
                              int K, int Ncols)
{
  int i = blockIdx.x * blockDim.x + threadIdx.x;
  if (i >= K * Ncols) return;
  size_t base = (size_t)blockIdx.y * K * Ncols;
  int k = i / Ncols, n = i - k * Ncols;
  int kt = k >> 5, rem = k & 31, kg = (rem >> 3) & 1, e = (rem & 7) + ((rem >> 4) << 3);
  size_t o = ((size_t)(kt * Ncols + n) * 2 + kg) * 16 + e;
  out[base + o] = to_bf16(in[base + i]);
}

// t = relu(x@W1+b1) in bf16 ; skip = x@Ws + bs  (K=2, scalar)
__global__ void embed_pre_kernel(const float* __restrict__ x,
                                 const float* __restrict__ W1, const float* __restrict__ b1,
                                 const float* __restrict__ Ws, const float* __restrict__ bs,
                                 bf16* __restrict__ tB, float* __restrict__ skip)
{
  int i = blockIdx.x * blockDim.x + threadIdx.x;   // N*128
  if (i >= N_NODES * HID) return;
  int n = i >> 7, c = i & 127;
  float x0 = x[(size_t)n * 2], x1 = x[(size_t)n * 2 + 1];
  float t = fmaxf(x0 * W1[c] + x1 * W1[HID + c] + b1[c], 0.f);
  tB[i]   = to_bf16(t);
  skip[i] = x0 * Ws[c] + x1 * Ws[HID + c] + bs[c];
}

// el/er = sum over head-dim of z * al / ar
__global__ void attn_coef_kernel(const float* __restrict__ z,
                                 const float* __restrict__ al, const float* __restrict__ ar,
                                 float* __restrict__ el, float* __restrict__ er)
{
  int i = blockIdx.x * blockDim.x + threadIdx.x;   // N*16
  if (i >= N_NODES * HEADS) return;
  int n = i >> 4, h = i & 15;
  const float* zp = z + (size_t)n * HID + h * DHD;
  const float* ap = al + h * DHD;
  const float* bp = ar + h * DHD;
  float a = 0.f, b = 0.f;
#pragma unroll
  for (int j = 0; j < DHD; ++j) { float zz = zp[j]; a += zz * ap[j]; b += zz * bp[j]; }
  el[i] = a; er[i] = b;
}

// e = leaky_relu(el[src] + er[dst]); segment max over dst via encoded atomicMax
__global__ void edge_max_kernel(const int* __restrict__ src, const int* __restrict__ dst,
                                const float* __restrict__ el, const float* __restrict__ er,
                                float* __restrict__ eatt, unsigned* __restrict__ mEnc)
{
  int i = blockIdx.x * blockDim.x + threadIdx.x;   // E*16
  if (i >= N_EDGES * HEADS) return;
  int e = i >> 4, h = i & 15;
  int s = src[e], d = dst[e];
  float v = el[s * HEADS + h] + er[d * HEADS + h];
  v = v > 0.f ? v : SLOPE * v;
  eatt[i] = v;
  atomicMax(&mEnc[d * HEADS + h], fenc(v));
}

// a = exp(e - m[dst]); denom += a; msg[dst] += a * z[src]
__global__ void edge_scatter_kernel(const int* __restrict__ src, const int* __restrict__ dst,
                                    const float* __restrict__ eatt, const unsigned* __restrict__ mEnc,
                                    const float* __restrict__ z,
                                    float* __restrict__ denom, float* __restrict__ msg)
{
  int i = blockIdx.x * blockDim.x + threadIdx.x;   // E*16
  if (i >= N_EDGES * HEADS) return;
  int e = i >> 4, h = i & 15;
  int s = src[e], d = dst[e];
  float m = fdec(mEnc[d * HEADS + h]);
  float a = expf(eatt[i] - m);
  atomicAdd(&denom[d * HEADS + h], a);
  const float* zp = z + (size_t)s * HID + h * DHD;
  float* mp = msg + (size_t)d * HID + h * DHD;
#pragma unroll
  for (int j = 0; j < DHD; ++j) atomicAdd(mp + j, a * zp[j]);
}

// s = h + msg/denom + bias ; accumulate per-channel BN sums
__global__ void gat_combine_kernel(const float* __restrict__ h, const float* __restrict__ msg,
                                   const float* __restrict__ denom, const float* __restrict__ bias,
                                   float* __restrict__ s,
                                   float* __restrict__ bsum, float* __restrict__ bsq)
{
  int c = threadIdx.x;                 // 128 channels
  int head = c >> 3;
  float a = 0.f, a2 = 0.f;
  for (int n = blockIdx.x; n < N_NODES; n += gridDim.x) {
    float den = denom[n * HEADS + head];
    den = den > 0.f ? den : 1.f;
    size_t idx = (size_t)n * HID + c;
    float v = h[idx] + msg[idx] / den + bias[c];
    s[idx] = v;
    a += v; a2 += v * v;
  }
  atomicAdd(&bsum[c], a);
  atomicAdd(&bsq[c], a2);
}

// per-channel BN stats only
__global__ void bn_stats_kernel(const float* __restrict__ s,
                                float* __restrict__ bsum, float* __restrict__ bsq)
{
  int c = threadIdx.x;
  float a = 0.f, a2 = 0.f;
  for (int n = blockIdx.x; n < N_NODES; n += gridDim.x) {
    float v = s[(size_t)n * HID + c];
    a += v; a2 += v * v;
  }
  atomicAdd(&bsum[c], a);
  atomicAdd(&bsq[c], a2);
}

// out = (s - mean) * rsqrt(var + eps) * g + b  (writes f32 and bf16)
__global__ void bn_apply_kernel(const float* __restrict__ s,
                                const float* __restrict__ bsum, const float* __restrict__ bsq,
                                const float* __restrict__ g, const float* __restrict__ b,
                                float* __restrict__ outF, bf16* __restrict__ outB)
{
  int i = blockIdx.x * blockDim.x + threadIdx.x;   // N*128
  if (i >= N_NODES * HID) return;
  int c = i & 127;
  const float invN = 1.f / (float)N_NODES;
  float mean = bsum[c] * invN;
  float var  = bsq[c] * invN - mean * mean;
  float r    = rsqrtf(var + BN_EPS);
  float v = (s[i] - mean) * r * g[c] + b[c];
  outF[i] = v;
  outB[i] = to_bf16(v);
}

// out[n] = dot(d1[n,:], W2[:,0]) + b2
__global__ void decoder_out_kernel(const float* __restrict__ d1, const float* __restrict__ W2,
                                   const float* __restrict__ b2, float* __restrict__ out)
{
  int n = blockIdx.x * blockDim.x + threadIdx.x;
  if (n >= N_NODES) return;
  const float4* row = (const float4*)(d1 + (size_t)n * HID);
  const float4* w   = (const float4*)W2;
  float acc = b2[0];
#pragma unroll
  for (int j = 0; j < HID / 4; ++j) {
    float4 a = row[j], ww = w[j];
    acc += a.x * ww.x + a.y * ww.y + a.z * ww.z + a.w * ww.w;
  }
  out[n] = acc;
}

// ---------------- host launch ----------------

static inline size_t align256(size_t x) { return (x + 255) & ~(size_t)255; }

extern "C" void kernel_launch(void* const* d_in, const int* in_sizes, int n_in,
                              void* d_out, int out_size, void* d_ws, size_t ws_size,
                              hipStream_t stream) {
  (void)in_sizes; (void)n_in; (void)out_size; (void)ws_size;
  const int N = N_NODES, E = N_EDGES;

  const float* x      = (const float*)d_in[0];
  const int*   src    = (const int*)d_in[1];
  const int*   dst    = (const int*)d_in[2];
  const float* emb_W1 = (const float*)d_in[3];
  const float* emb_b1 = (const float*)d_in[4];
  const float* emb_W2 = (const float*)d_in[5];
  const float* emb_b2 = (const float*)d_in[6];
  const float* emb_Ws = (const float*)d_in[7];
  const float* emb_bs = (const float*)d_in[8];
  const float* gat_W  = (const float*)d_in[9];
  const float* gat_al = (const float*)d_in[10];
  const float* gat_ar = (const float*)d_in[11];
  const float* gat_b  = (const float*)d_in[12];
  const float* bn1_g  = (const float*)d_in[13];
  const float* bn1_b  = (const float*)d_in[14];
  const float* ff_W1  = (const float*)d_in[15];
  const float* ff_b1  = (const float*)d_in[16];
  const float* ff_W2  = (const float*)d_in[17];
  const float* ff_b2  = (const float*)d_in[18];
  const float* bn2_g  = (const float*)d_in[19];
  const float* bn2_b  = (const float*)d_in[20];
  const float* dec_W1 = (const float*)d_in[21];
  const float* dec_b1 = (const float*)d_in[22];
  const float* dec_W2 = (const float*)d_in[23];
  const float* dec_b2 = (const float*)d_in[24];

  // ---- carve workspace ----
  char* w = (char*)d_ws;
  auto carve = [&](size_t bytes) { void* p = (void*)w; w += align256(bytes); return p; };

  float*    xsF   = (float*)carve((size_t)(LAYERS + 1) * N * HID * 4); // h after each stage
  bf16*     xsB   = (bf16*) carve((size_t)(LAYERS + 1) * N * HID * 2);
  bf16*     uB    = (bf16*) carve((size_t)N * FF_DIM * 2);             // FF hidden (also embed t)
  float*    zF    = (float*)carve((size_t)N * HID * 4);                // GAT z (also decoder d1)
  float*    hmidF = (float*)carve((size_t)N * HID * 4);                // BN1 output
  bf16*     hmidB = (bf16*) carve((size_t)N * HID * 2);
  float*    sF    = (float*)carve((size_t)N * HID * 4);                // pre-BN sum (also embed skip)
  float*    msgF  = (float*)carve((size_t)N * HID * 4);
  float*    elF   = (float*)carve((size_t)N * HEADS * 4);
  float*    erF   = (float*)carve((size_t)N * HEADS * 4);
  float*    denF  = (float*)carve((size_t)N * HEADS * 4);
  unsigned* mEnc  = (unsigned*)carve((size_t)N * HEADS * 4);
  float*    eatt  = (float*)carve((size_t)E * HEADS * 4);
  float*    bsum  = (float*)carve(HID * 4);
  float*    bsq   = (float*)carve(HID * 4);
  bf16*     embW2B = (bf16*)carve((size_t)HID * HID * 2);
  bf16*     gatWB  = (bf16*)carve((size_t)LAYERS * HID * HID * 2);
  bf16*     ffW1B  = (bf16*)carve((size_t)LAYERS * HID * FF_DIM * 2);
  bf16*     ffW2B  = (bf16*)carve((size_t)LAYERS * FF_DIM * HID * 2);
  bf16*     decW1B = (bf16*)carve((size_t)(LAYERS + 1) * HID * HID * 2);

  const int T = 256;
  auto cdiv = [](int a, int b) { return (a + b - 1) / b; };

  // ---- weight conversion + fragment packing (f32 -> packed bf16) ----
  pack_b_kernel<<<dim3(cdiv(HID * HID, T), 1), T, 0, stream>>>(emb_W2, embW2B, HID, HID);
  pack_b_kernel<<<dim3(cdiv(HID * HID, T), LAYERS), T, 0, stream>>>(gat_W, gatWB, HID, HID);
  pack_b_kernel<<<dim3(cdiv(HID * FF_DIM, T), LAYERS), T, 0, stream>>>(ff_W1, ffW1B, HID, FF_DIM);
  pack_b_kernel<<<dim3(cdiv(FF_DIM * HID, T), LAYERS), T, 0, stream>>>(ff_W2, ffW2B, FF_DIM, HID);
  pack_b_kernel<<<dim3(cdiv(HID * HID, T), LAYERS + 1), T, 0, stream>>>(dec_W1, decW1B, HID, HID);

  // ---- embed: h0 = relu(x@W1+b1)@W2 + b2 + (x@Ws + bs) ----
  embed_pre_kernel<<<cdiv(N * HID, T), T, 0, stream>>>(x, emb_W1, emb_b1, emb_Ws, emb_bs, uB, sF);
  gemm_wmma<HID><<<dim3(N / 32, 1), T, 0, stream>>>(uB, embW2B, emb_b2, sF, xsF, xsB, HID, 0);

  // ---- GAT layers ----
  for (int l = 0; l < LAYERS; ++l) {
    const float* hF = xsF + (size_t)l * N * HID;
    const bf16*  hB = xsB + (size_t)l * N * HID;

    // z = h @ W_l
    gemm_wmma<HID><<<dim3(N / 32, 1), T, 0, stream>>>(hB, gatWB + (size_t)l * HID * HID,
                                                      nullptr, nullptr, zF, nullptr, HID, 0);
    attn_coef_kernel<<<cdiv(N * HEADS, T), T, 0, stream>>>(zF, gat_al + l * HEADS * DHD,
                                                           gat_ar + l * HEADS * DHD, elF, erF);
    hipMemsetAsync(mEnc, 0, (size_t)N * HEADS * 4, stream);   // encoded -inf floor
    hipMemsetAsync(denF, 0, (size_t)N * HEADS * 4, stream);
    hipMemsetAsync(msgF, 0, (size_t)N * HID * 4, stream);
    edge_max_kernel<<<cdiv(E * HEADS, T), T, 0, stream>>>(src, dst, elF, erF, eatt, mEnc);
    edge_scatter_kernel<<<cdiv(E * HEADS, T), T, 0, stream>>>(src, dst, eatt, mEnc, zF, denF, msgF);

    // s = h + msg/denom + bias ; BN1
    hipMemsetAsync(bsum, 0, HID * 4, stream);
    hipMemsetAsync(bsq, 0, HID * 4, stream);
    gat_combine_kernel<<<256, HID, 0, stream>>>(hF, msgF, denF, gat_b + l * HID, sF, bsum, bsq);
    bn_apply_kernel<<<cdiv(N * HID, T), T, 0, stream>>>(sF, bsum, bsq,
                                                        bn1_g + l * HID, bn1_b + l * HID,
                                                        hmidF, hmidB);

    // FF: u = relu(h@W1+b1) ; s = u@W2 + b2 + h ; BN2
    gemm_wmma<HID><<<dim3(N / 32, 2), T, 0, stream>>>(hmidB, ffW1B + (size_t)l * HID * FF_DIM,
                                                      ff_b1 + l * FF_DIM, nullptr,
                                                      nullptr, uB, FF_DIM, 1);
    gemm_wmma<FF_DIM><<<dim3(N / 32, 1), T, 0, stream>>>(uB, ffW2B + (size_t)l * FF_DIM * HID,
                                                         ff_b2 + l * HID, hmidF,
                                                         sF, nullptr, HID, 0);
    hipMemsetAsync(bsum, 0, HID * 4, stream);
    hipMemsetAsync(bsq, 0, HID * 4, stream);
    bn_stats_kernel<<<256, HID, 0, stream>>>(sF, bsum, bsq);
    bn_apply_kernel<<<cdiv(N * HID, T), T, 0, stream>>>(sF, bsum, bsq,
                                                        bn2_g + l * HID, bn2_b + l * HID,
                                                        xsF + (size_t)(l + 1) * N * HID,
                                                        xsB + (size_t)(l + 1) * N * HID);
  }

  // ---- decoder: out = relu(concat(xs)@W1 + b1) @ W2 + b2 ----
  gemm_dec_wmma<<<N / 32, T, 0, stream>>>(xsB, decW1B, dec_b1, zF /* d1 */);
  decoder_out_kernel<<<cdiv(N, T), T, 0, stream>>>(zF, dec_W2, dec_b2, (float*)d_out);
}
